// Nessler2009_90683939488507
// MI455X (gfx1250) — matrix operand order repaired and checked by reading, not compiled, not run on hardware
//
#include <hip/hip_runtime.h>
#include <math.h>
#include <stdint.h>

// ---------------- problem constants ----------------
#define NB    256       // batch
#define NT    200       // time steps
#define NIN   1568      // input neurons
#define NO    100       // output neurons
#define OPAD  112       // O padded to 7 WMMA tiles
#define KPAD  128       // per-WG input rows padded to 4 K-chunks of 32
#define NWG   16        // workgroups (one cluster)
#define INW   98        // NIN / NWG rows owned per WG
#define LRATE 0.001f

// workspace layout (floats)
#define LM_OFF   0
#define LOG_OFF  (NIN*OPAD)                 // 2 ping-pong logits buffers
#define PRI_OFF  (LOG_OFF + 2*NB*OPAD)
#define CTR_OFF  (PRI_OFF + OPAD)
#define WIDX_OFF (CTR_OFF + 16)
#define WS_FLOATS (WIDX_OFF + NB)

typedef _Float16 v16h __attribute__((ext_vector_type(16)));
typedef float    v8f  __attribute__((ext_vector_type(8)));
typedef unsigned u32x4 __attribute__((ext_vector_type(4)));
typedef int      i32x8 __attribute__((ext_vector_type(8)));
typedef int      i32x4 __attribute__((ext_vector_type(4)));

__device__ __forceinline__ v8f wmma16(v16h a, v16h b, v8f c) {
    return __builtin_amdgcn_wmma_f32_16x16x32_f16(false, a, false, b, (short)0, c,
                                                  false, false);
}

// A-matrix 16x32 f16 VGPR layout (ISA 7.12.2): lane holds row M=lane%16;
// element e of the 16-wide fragment maps to K = kmapA(e, lane/16).
__device__ __forceinline__ int kmapA(int e, int hf) {
    int p = e >> 1, s = e & 1;
    return ((p & 3) << 1) + s + (hf << 3) + ((p >> 2) << 4);
}

__device__ __forceinline__ float hash_uniform(unsigned t, unsigned b) {
    unsigned x = t * 0x9E3779B9u + b * 0x85EBCA6Bu + 0x27220A95u;
    x ^= x >> 16; x *= 0x7FEB352Du;
    x ^= x >> 15; x *= 0x846CA68Bu;
    x ^= x >> 16;
    return (float)(x >> 8) * (1.0f / 16777216.0f);
}

// Cluster-wide sync: hardware cluster barrier (signal+wait, wave0) plus a
// monotone atomic barrier so the kernel is also correct without cluster launch.
__device__ void cluster_sync(unsigned* ctr) {
    __threadfence();
    __syncthreads();
    if (threadIdx.x == 0) {
#if __has_builtin(__builtin_amdgcn_s_cluster_barrier)
        __builtin_amdgcn_s_cluster_barrier();
#endif
        unsigned old = atomicAdd(ctr, 1u);
        unsigned target = (old / NWG + 1u) * NWG;
        while (atomicAdd(ctr, 0u) < target) { __builtin_amdgcn_s_sleep(1); }
    }
    __syncthreads();
}

// TDM: DMA the 2D tile x[brow0..brow0+127, t, wg*INW .. +INW) into LDS.
// Row stride between batch rows is NT*NIN elements; innermost dim contiguous.
__device__ __forceinline__ void tdm_load_slice(const int* x, unsigned lds_addr,
                                               int t, int brow0, int wg) {
    unsigned long long ga = (unsigned long long)(uintptr_t)x +
        4ull * ((unsigned long long)((unsigned long long)brow0 * NT + (unsigned)t) * NIN
                + (unsigned)(wg * INW));
    unsigned long long stride0 = (unsigned long long)NT * NIN;   // dim0 stride
#if __has_builtin(__builtin_amdgcn_tensor_load_to_lds)
    u32x4 g0;
    g0[0] = 1u;                                           // count=1 valid D#
    g0[1] = lds_addr;                                     // LDS byte address
    g0[2] = (unsigned)(ga & 0xffffffffull);               // global_addr[31:0]
    g0[3] = (unsigned)((ga >> 32) & 0x1ffffffull) | (2u << 30);  // addr hi | type=2
    i32x8 g1;
    g1[0] = (int)(2u << 16);                              // data_size = 4B
    g1[1] = (int)(((unsigned)INW & 0xffffu) << 16);       // tensor_dim0 lo
    g1[2] = (int)((((unsigned)INW >> 16) & 0xffffu) | (128u << 16)); // dim0 hi|dim1 lo
    g1[3] = (int)(0u | (((unsigned)INW & 0xffffu) << 16));           // dim1 hi|tile_dim0
    g1[4] = (int)(128u);                                  // tile_dim1 = 128 rows
    g1[5] = (int)(stride0 & 0xffffffffull);               // tensor_dim0_stride lo
    g1[6] = (int)((stride0 >> 32) & 0xffffull);           // stride hi | dim1_stride=0
    g1[7] = 0;
    i32x4 z4 = {0, 0, 0, 0};                              // groups 2/3 unused (2D)
    i32x8 z8 = {0, 0, 0, 0, 0, 0, 0, 0};
    __builtin_amdgcn_tensor_load_to_lds(g0, g1, z4, z4, z8, 0);
    __builtin_amdgcn_s_wait_tensorcnt(0);
#else
    (void)ga; (void)stride0; (void)lds_addr;
#endif
}

__global__ __launch_bounds__(256, 1)
void nessler_cdna5(const int* __restrict__ x, const float* __restrict__ Lin,
                   const float* __restrict__ Pin, float* __restrict__ out,
                   float* __restrict__ ws) {
    __shared__ _Float16      sL[KPAD * OPAD];        // 28 KB  B-matrix tile
    __shared__ unsigned short sMask[NB * INW];       // 50 KB  11-bit spike windows
    __shared__ int           sStage[128 * INW];      // 50 KB  TDM staging (half batch)
    __shared__ unsigned      sFired[NB * 4];         // 4 KB   ever-fired bitset (B x O)
    __shared__ int           sWidx[NB];              // 1 KB

    float*    Lm    = ws + LM_OFF;          // padded master L (NIN x OPAD), f32
    float*    gLog  = ws + LOG_OFF;         // 2 x (NB x OPAD) logits buffers
    float*    gPri  = ws + PRI_OFF;         // padded prior
    unsigned* ctr   = (unsigned*)(ws + CTR_OFF);
    int*      gWidx = (int*)(ws + WIDX_OFF);

    const int tid  = threadIdx.x;
    const int wg   = blockIdx.x;
    const int lane = tid & 31;
    const int wid  = tid >> 5;
    const int hf   = lane >> 4;
    const int l16  = lane & 15;
    const unsigned stageAddr = (unsigned)(size_t)&sStage[0];

    // ---------------- init ----------------
    for (int i = tid; i < NB * INW; i += 256) sMask[i] = 0;
    for (int i = tid; i < NB * 4;   i += 256) sFired[i] = 0u;
    for (int idx = tid; idx < INW * OPAD; idx += 256) {
        int il = idx / OPAD, c = idx % OPAD;
        Lm[(wg * INW + il) * OPAD + c] = (c < NO) ? Lin[(wg * INW + il) * NO + c] : 0.f;
    }
    if (wg == 0) {
        if (tid < OPAD) gPri[tid] = (tid < NO) ? Pin[tid] : 0.f;
        __syncthreads();
        if (tid == 0) {                              // prior: clamp + logsumexp
            float m = -1e30f;
            for (int o = 0; o < NO; ++o) {
                float p = fminf(fmaxf(gPri[o], -5.f), 0.f);
                gPri[o] = p; m = fmaxf(m, p);
            }
            float s = 0.f;
            for (int o = 0; o < NO; ++o) s += expf(gPri[o] - m);
            float lse = m + logf(s);
            for (int o = 0; o < NO; ++o) gPri[o] -= lse;
        }
    }
    cluster_sync(ctr);

    // initial population-pair normalization of L (rows <784 live in WGs 0..7)
    if (wg < 8) {
        for (int idx = tid; idx < INW * NO; idx += 256) {
            int il = idx / NO, c = idx % NO;
            int i = wg * INW + il;
            float a = fminf(fmaxf(Lm[i * OPAD + c], -5.f), 0.f);
            float b = fminf(fmaxf(Lm[(i + 784) * OPAD + c], -5.f), 0.f);
            float m = fmaxf(a, b);
            float lse = m + logf(expf(a - m) + expf(b - m));
            Lm[i * OPAD + c] = a - lse;
            Lm[(i + 784) * OPAD + c] = b - lse;
        }
    }
    cluster_sync(ctr);

    // load f16 L tile into LDS
    for (int idx = tid; idx < KPAD * OPAD; idx += 256) {
        int k = idx / OPAD, n = idx % OPAD;
        float v = (k < INW && n < NO) ? Lm[(wg * INW + k) * OPAD + n] : 0.f;
        sL[idx] = (_Float16)v;
    }
    __syncthreads();

    // ---------------- time loop ----------------
    for (int t = 0; t < NT; ++t) {
        // -- phase 1: TDM-stage x[:,t,:] and update 11-bit spike windows --
        for (int h2 = 0; h2 < 2; ++h2) {
            if (wid == 0) {                       // one wave issues the DMA
                tdm_load_slice(x, stageAddr, t, h2 * 128, wg);
                if (t + 1 < NT && lane == 0)
                    __builtin_prefetch(x + ((size_t)(h2 * 128) * NT + t + 1) * NIN
                                         + wg * INW, 0, 1);
            }
            __syncthreads();
            for (int idx = tid; idx < 128 * INW; idx += 256) {
                int b = h2 * 128 + idx / INW, il = idx % INW;
                unsigned m = sMask[b * INW + il];
                m = ((m << 1) | ((unsigned)sStage[idx] & 1u)) & 0x7FFu;
                sMask[b * INW + il] = (unsigned short)m;
            }
            __syncthreads();
        }

        // -- phase 2: partial logits GEMM, pot(B x 128) @ L(128 x OPAD) --
        float* logCur = gLog + (t & 1) * NB * OPAD;
        for (int tile = wid; tile < 16 * 7; tile += 8) {
            int mt = tile / 7, nt = tile % 7;
            v8f acc = {0.f, 0.f, 0.f, 0.f, 0.f, 0.f, 0.f, 0.f};
            for (int kt = 0; kt < 4; ++kt) {
                union { _Float16 h[16]; v16h v; } A, Bf;
                int brow = mt * 16 + l16;
                for (int e = 0; e < 16; ++e) {
                    int kk = kt * 32 + kmapA(e, hf);
                    _Float16 p = (_Float16)0.f;
                    if (kk < INW && (sMask[brow * INW + kk] & 0x3FFu))
                        p = (_Float16)1.f;              // pot: 10-step OR window
                    A.h[e] = p;
                }
                int krow = kt * 32;
                for (int e = 0; e < 16; ++e)
                    Bf.h[e] = sL[(krow + kmapA(e, hf)) * OPAD + nt * 16 + l16];
                acc = wmma16(A.v, Bf.v, acc);
            }
            for (int v = 0; v < 8; ++v) {
                int row = mt * 16 + v + 8 * hf;
                int col = nt * 16 + l16;
                atomicAdd(&logCur[row * OPAD + col], acc[v]);
            }
        }
        cluster_sync(ctr);

        // -- phase 3: WG0 samples categorical(exp(exp(logits+prior))) --
        if (wg == 0) {
            int b = tid;                                   // 256 threads = 256 rows
            float zmax = -1e30f;
            for (int o = 0; o < NO; ++o)
                zmax = fmaxf(zmax, expf(logCur[b * OPAD + o] + gPri[o]));
            float tot = 0.f;
            for (int o = 0; o < NO; ++o)
                tot += expf(expf(logCur[b * OPAD + o] + gPri[o]) - zmax);
            float u = hash_uniform((unsigned)t, (unsigned)b) * tot;
            int pick = NO - 1; float csum = 0.f;
            for (int o = 0; o < NO; ++o) {
                csum += expf(expf(logCur[b * OPAD + o] + gPri[o]) - zmax);
                if (u < csum) { pick = o; break; }
            }
            gWidx[b] = pick;
            sWidx[b] = pick;
            __syncthreads();
            if (tid < NO) {                                // prior STDP update
                int cnt = 0;
                for (int b2 = 0; b2 < NB; ++b2) cnt += (sWidx[b2] == tid);
                float wm = (float)cnt / (float)NB;
                float p = gPri[tid];
                gPri[tid] = p + (LRATE / (float)(t + 1)) *
                            ((-5.f * p - 1.f) * wm - (1.f - wm));
            }
            __syncthreads();
            if (tid == 0) {                                // prior normalize
                float m = -1e30f;
                for (int o = 0; o < NO; ++o) {
                    float p = fminf(fmaxf(gPri[o], -5.f), 0.f);
                    gPri[o] = p; m = fmaxf(m, p);
                }
                float s = 0.f;
                for (int o = 0; o < NO; ++o) s += expf(gPri[o] - m);
                float lse = m + logf(s);
                for (int o = 0; o < NO; ++o) gPri[o] -= lse;
            }
            __syncthreads();
            for (int idx = tid; idx < NB * OPAD; idx += 256)  // re-zero for t+2
                logCur[idx] = 0.f;
        }
        cluster_sync(ctr);

        // -- phase 4: outer products via WMMA (A^T traces x win/post) + L update --
        sWidx[tid] = gWidx[tid];
        __syncthreads();
        {   // ever-fired bitset includes the current step (tpost semantics)
            int o = sWidx[tid];
            sFired[tid * 4 + (o >> 5)] |= 1u << (o & 31);
        }
        __syncthreads();

        float inv = LRATE / (float)(t + 1);
        for (int tile = wid; tile < 7 * 7; tile += 8) {
            int it = tile / 7, ot = tile % 7;
            v8f aP = {0.f,0.f,0.f,0.f,0.f,0.f,0.f,0.f};   // LTP^T . win
            v8f aQ = aP;                                   // LTD^T . win
            v8f aR = aP;                                   // pot^T . post
            for (int kt = 0; kt < 8; ++kt) {
                union { _Float16 h[16]; v16h v; } Altp, Altd, Apot, Bw, Bp;
                int il = it * 16 + l16;
                for (int e = 0; e < 16; ++e) {
                    int bb = kt * 32 + kmapA(e, hf);
                    unsigned m = (il < INW) ? (unsigned)sMask[bb * INW + il] : 0u;
                    _Float16 one = (_Float16)1.f, zero = (_Float16)0.f;
                    Altp.h[e] = (m & 0x7FFu) ? one : zero;           // 11-step window
                    Altd.h[e] = (il < INW && !(m & 0x7FFu)) ? one : zero;
                    Apot.h[e] = (m & 0x3FFu) ? one : zero;
                }
                int col = ot * 16 + l16;
                for (int e = 0; e < 16; ++e) {
                    int bb = kt * 32 + kmapA(e, hf);
                    int wsel = sWidx[bb];
                    bool win = (wsel == col);
                    bool fired = (sFired[bb * 4 + (col >> 5)] >> (col & 31)) & 1u;
                    Bw.h[e] = win ? (_Float16)1.f : (_Float16)0.f;
                    Bp.h[e] = (fired && !win) ? (_Float16)1.f : (_Float16)0.f;
                }
                aP = wmma16(Altp.v, Bw.v, aP);
                aQ = wmma16(Altd.v, Bw.v, aQ);
                aR = wmma16(Apot.v, Bp.v, aR);
            }
            for (int v = 0; v < 8; ++v) {
                int il = it * 16 + v + 8 * hf;
                int col = ot * 16 + l16;
                if (il < INW && col < NO) {
                    int gi = (wg * INW + il) * OPAD + col;
                    float Lv = Lm[gi];
                    float mltp = aP[v] * (1.f / NB);
                    float mltd = aQ[v] * (1.f / NB);
                    float mpp  = aR[v] * (1.f / NB);
                    float dw = (5.f * expf(-Lv) - 1.f) * mltp + mltd - mpp;
                    Lm[gi] = Lv + inv * dw;
                }
            }
        }
        cluster_sync(ctr);

        // -- phase 5: clamp + population-pair logsumexp normalize --
        if (wg < 8) {
            for (int idx = tid; idx < INW * NO; idx += 256) {
                int il = idx / NO, c = idx % NO;
                int i = wg * INW + il;
                float a = fminf(fmaxf(Lm[i * OPAD + c], -5.f), 0.f);
                float b = fminf(fmaxf(Lm[(i + 784) * OPAD + c], -5.f), 0.f);
                float m = fmaxf(a, b);
                float lse = m + logf(expf(a - m) + expf(b - m));
                Lm[i * OPAD + c] = a - lse;
                Lm[(i + 784) * OPAD + c] = b - lse;
            }
        }
        cluster_sync(ctr);

        // refresh this WG's f16 L tile for the next step's GEMM
        for (int idx = tid; idx < KPAD * OPAD; idx += 256) {
            int k = idx / OPAD, n = idx % OPAD;
            float v = (k < INW && n < NO) ? Lm[(wg * INW + k) * OPAD + n] : 0.f;
            sL[idx] = (_Float16)v;
        }
        __syncthreads();
    }

    // ---------------- epilogue: unpad to output ----------------
    for (int idx = tid; idx < INW * NO; idx += 256) {
        int il = idx / NO, c = idx % NO;
        out[(wg * INW + il) * NO + c] = Lm[(wg * INW + il) * OPAD + c];
    }
    if (wg == 0 && tid < NO) out[NIN * NO + tid] = gPri[tid];
}

extern "C" void kernel_launch(void* const* d_in, const int* in_sizes, int n_in,
                              void* d_out, int out_size, void* d_ws, size_t ws_size,
                              hipStream_t stream) {
    (void)in_sizes; (void)n_in; (void)out_size; (void)ws_size;
    const int*   x   = (const int*)d_in[0];
    const float* Lin = (const float*)d_in[1];
    const float* Pin = (const float*)d_in[2];
    float* out = (float*)d_out;
    float* ws  = (float*)d_ws;

    // zero workspace (logits buffers + barrier counter need zeros; capture-safe)
    (void)hipMemsetAsync(d_ws, 0, (size_t)WS_FLOATS * sizeof(float), stream);

    // one persistent 16-workgroup cluster; 256 threads = 8 wave32 per WGP
    nessler_cdna5<<<dim3(NWG), dim3(256), 0, stream>>>(x, Lin, Pin, out, ws);
}
